// Model_42442866819683
// MI455X (gfx1250) — matrix operand order repaired and checked by reading, not compiled, not run on hardware
//
#include <hip/hip_runtime.h>
#include <hip/hip_bf16.h>
#include <math.h>

#define DEVI __device__ __forceinline__

// ---------------- problem constants ----------------
constexpr int  BB  = 2;
constexpr int  CC  = 128;
constexpr int  HXc = 64, HYc = 64, HZc = 32;
constexpr long NNc   = (long)HXc * HYc * HZc;   // 131072
constexpr long ROWSc = (long)BB * NNc;          // 262144
constexpr long BNCc  = ROWSc * CC;              // 33,554,432 elements
constexpr int  MM = 64, NHc = 4, DHc = 32;
constexpr int  GG = 32;
constexpr float ATTN_SCALE = 0.17677669529663687f; // 1/sqrt(32)

typedef unsigned short u16;
typedef __bf16 bf16_t;
typedef bf16_t v16bf __attribute__((ext_vector_type(16)));
typedef float  v8f   __attribute__((ext_vector_type(8)));

union FragBF { v16bf v; u16 u[16]; };

DEVI u16 f2bf(float f) {
  union { float f; unsigned u; } x; x.f = f;
  unsigned r = x.u + 0x7FFFu + ((x.u >> 16) & 1u);   // RNE
  return (u16)(r >> 16);
}

// ---------------- WMMA fragment loaders (contiguous k = 16*(l>>4) + s order, wave32) ----------------
// A: 16x32 bf16. lane l: row m = m0 + (l&15); VGPR v holds K pair:
//   k = 2v + (l>>4)*16
DEVI v16bf loadA_bf(const u16* __restrict__ A, int lda, int m0, int k0) {
  FragBF fr; int l = threadIdx.x & 31;
  const u16* row = A + (size_t)(m0 + (l & 15)) * lda + k0;
  int half = (l >> 4) * 16;
#pragma unroll
  for (int v = 0; v < 8; ++v) {
    int kb = 2 * v + half;
    fr.u[2 * v]     = row[kb];
    fr.u[2 * v + 1] = row[kb + 1];
  }
  return fr.v;
}
DEVI v16bf loadA_f32(const float* __restrict__ A, int lda, int m0, int k0, float s) {
  FragBF fr; int l = threadIdx.x & 31;
  const float* row = A + (size_t)(m0 + (l & 15)) * lda + k0;
  int half = (l >> 4) * 16;
#pragma unroll
  for (int v = 0; v < 8; ++v) {
    int kb = 2 * v + half;
    fr.u[2 * v]     = f2bf(row[kb] * s);
    fr.u[2 * v + 1] = f2bf(row[kb + 1] * s);
  }
  return fr.v;
}
// B: 32x16 bf16, B[k][n] = base[k*kstride + n]. lane l: n = n0+(l&15),
// lanes 0-15 hold K 0-15, lanes 16-31 hold K 16-31.
DEVI v16bf loadB_k(const u16* __restrict__ base, size_t kstride, int k0, int n0) {
  FragBF fr; int l = threadIdx.x & 31;
  int n = n0 + (l & 15);
  int kh = (l >> 4) * 16;
#pragma unroll
  for (int e = 0; e < 16; ++e)
    fr.u[e] = base[(size_t)(k0 + kh + e) * kstride + n];
  return fr.v;
}
// B[k][n] = base[n*nstride + k] (row-major W[o][c] used directly as B^T);
// each lane reads 16 contiguous bf16 -> vectorizes to 2x b128.
DEVI v16bf loadB_n(const u16* __restrict__ base, size_t nstride, int k0, int n0) {
  FragBF fr; int l = threadIdx.x & 31;
  size_t j = (size_t)(n0 + (l & 15));
  const u16* p = base + j * nstride + k0 + (l >> 4) * 16;
#pragma unroll
  for (int e = 0; e < 16; ++e) fr.u[e] = p[e];
  return fr.v;
}
DEVI v8f wmma_bf16(v16bf a, v16bf b, v8f c) {
  return __builtin_amdgcn_wmma_f32_16x16x32_bf16(false, a, false, b, (short)0, c, false, false);
}

// ---------------- generic 128-wide WMMA GEMM ----------------
// D(Mrows x Ncols) = A(Mrows x Kdim) * W^T [+bias], W row-major [n][k] bf16.
// Epilogue MODE: 0: dstF = beta*dstF + acc + bias   1: dstBF = bf16(acc + bias)
//                2: dstF = aux1 + (*gscale)*sigmoid(acc+bias)*aux2
template <bool AF32, int MODE>
__global__ __launch_bounds__(256) void gemm_kernel(
    const void* __restrict__ Asrc, int lda, float ascale,
    const u16* __restrict__ Wbf, int ldb, int Kdim,
    const float* __restrict__ bias, int Ncols, float beta,
    float* __restrict__ dstF, u16* __restrict__ dstBF, int ldd,
    const float* __restrict__ gscale, const float* __restrict__ aux1,
    const float* __restrict__ aux2, long Mrows)
{
  long row0 = (long)blockIdx.x * 128;
  int rows = (int)((Mrows - row0) > 128 ? 128 : (Mrows - row0));
  int mt_n = rows >> 4, nt_n = Ncols >> 4;
  int wave = threadIdx.x >> 5, l = threadIdx.x & 31;
  int ntiles = mt_n * nt_n;
  for (int t = wave; t < ntiles; t += 8) {
    int mt = t / nt_n, nt = t % nt_n;
    // prefetch next tile's A rows into cache while WMMAs issue
    if (t + 8 < ntiles) {
      int mt2 = (t + 8) / nt_n;
      const char* pf = (const char*)Asrc +
          (size_t)(row0 + mt2 * 16 + (l & 15)) * lda * (AF32 ? 4 : 2);
      __builtin_prefetch(pf, 0, 1);
    }
    v8f acc = {};
    for (int kk = 0; kk < Kdim; kk += 32) {
      v16bf a;
      if (AF32) a = loadA_f32((const float*)Asrc + row0 * lda, lda, mt * 16, kk, ascale);
      else      a = loadA_bf((const u16*)Asrc + row0 * lda, lda, mt * 16, kk);
      v16bf b = loadB_n(Wbf, (size_t)ldb, kk, nt * 16);
      acc = wmma_bf16(a, b, acc);
    }
    int col = nt * 16 + (l & 15);
    float bv = bias ? bias[col] : 0.f;
#pragma unroll
    for (int v = 0; v < 8; ++v) {
      long r = row0 + mt * 16 + v + (l >> 4) * 8;
      size_t di = (size_t)r * ldd + col;
      float x = acc[v];
      if (MODE == 0) {
        float prev = (beta != 0.f) ? dstF[di] * beta : 0.f;
        dstF[di] = prev + x + bv;
      } else if (MODE == 1) {
        dstBF[di] = f2bf(x + bv);
      } else {
        float g = 1.f / (1.f + __expf(-(x + bv)));
        dstF[di] = aux1[di] + (*gscale) * g * aux2[di];
      }
    }
  }
}

// ---------------- depthwise 3x3x3 conv ----------------
__global__ __launch_bounds__(256) void conv_dw_kernel(
    const float* __restrict__ h, const float* __restrict__ w, float* __restrict__ y)
{
  long idx = (long)blockIdx.x * 256 + threadIdx.x;
  int c = (int)(idx & 127);
  long bn = idx >> 7;
  int n = (int)(bn % NNc), b = (int)(bn / NNc);
  int x = n & (HXc - 1), yy = (n >> 6) & (HYc - 1), z = n >> 12;
  const float* wc = w + c * 27;
  float acc = 0.f;
#pragma unroll
  for (int dz = -1; dz <= 1; ++dz) {
    int zz = z + dz; if ((unsigned)zz >= (unsigned)HZc) continue;
#pragma unroll
    for (int dy = -1; dy <= 1; ++dy) {
      int yn = yy + dy; if ((unsigned)yn >= (unsigned)HYc) continue;
#pragma unroll
      for (int dx = -1; dx <= 1; ++dx) {
        int xn = x + dx; if ((unsigned)xn >= (unsigned)HXc) continue;
        long n2 = ((long)zz * HYc + yn) * HXc + xn;
        acc += h[((size_t)b * NNc + n2) * CC + c] * wc[(dx + 1) * 9 + (dy + 1) * 3 + (dz + 1)];
      }
    }
  }
  y[idx] = acc;
}

// ---------------- group-norm stats ----------------
__global__ __launch_bounds__(256) void gn_stats_kernel(
    const float* __restrict__ y, float* __restrict__ ssum, float* __restrict__ ssq)
{
  __shared__ float s1[256], s2[256];
  long R0 = (long)blockIdx.x * 128;
  int b = (int)(R0 / NNc);
  int c = threadIdx.x & 127, rh = threadIdx.x >> 7;
  float a = 0.f, q = 0.f;
  for (int i = 0; i < 64; ++i) {
    float v = y[(R0 + rh + 2 * i) * CC + c];
    a += v; q += v * v;
  }
  s1[threadIdx.x] = a; s2[threadIdx.x] = q;
  __syncthreads();
  if (threadIdx.x < 128) { s1[threadIdx.x] += s1[threadIdx.x + 128]; s2[threadIdx.x] += s2[threadIdx.x + 128]; }
  __syncthreads();
  if (threadIdx.x < 128 && (threadIdx.x & 3) == 0) {
    float A = s1[threadIdx.x] + s1[threadIdx.x + 1] + s1[threadIdx.x + 2] + s1[threadIdx.x + 3];
    float Q = s2[threadIdx.x] + s2[threadIdx.x + 1] + s2[threadIdx.x + 2] + s2[threadIdx.x + 3];
    int g = threadIdx.x >> 2;
    atomicAdd(&ssum[b * GG + g], A);
    atomicAdd(&ssq[b * GG + g], Q);
  }
}

__global__ void gn_final_kernel(const float* ssum, const float* ssq, float* mu, float* rs)
{
  int i = threadIdx.x;
  if (i < BB * GG) {
    float cnt = (float)((CC / GG) * NNc);
    float m = ssum[i] / cnt;
    float v = ssq[i] / cnt - m * m;
    mu[i] = m;
    rs[i] = rsqrtf(v + 1e-5f);
  }
}

// ---------------- GN affine + exact GELU -> bf16 ----------------
__global__ __launch_bounds__(256) void gn_gelu_kernel(
    const float* __restrict__ y, const float* __restrict__ mu, const float* __restrict__ rs,
    const float* __restrict__ gw, const float* __restrict__ gb, u16* __restrict__ dst)
{
  long idx = (long)blockIdx.x * 256 + threadIdx.x;
  int c = (int)(idx & 127);
  int b = (int)((idx >> 7) / NNc);
  int g = c >> 2;
  float v = (y[idx] - mu[b * GG + g]) * rs[b * GG + g] * gw[c] + gb[c];
  float ge = 0.5f * v * (1.f + erff(v * 0.7071067811865475f));
  dst[idx] = f2bf(ge);
}

__global__ __launch_bounds__(256) void cvt_bf16_kernel(const float* __restrict__ s, u16* __restrict__ d)
{
  long idx = (long)blockIdx.x * 256 + threadIdx.x;
  d[idx] = f2bf(s[idx]);
}

// ---------------- wave-per-row LayerNorm -> bf16 ----------------
__global__ __launch_bounds__(256) void ln_rows_kernel(
    const float* __restrict__ src, const float* __restrict__ w, const float* __restrict__ b,
    u16* __restrict__ dst, long rows)
{
  int wave = threadIdx.x >> 5, l = threadIdx.x & 31;
  long r = (long)blockIdx.x * 8 + wave;
  if (r >= rows) return;
  const float4* rp = (const float4*)(src + r * CC);
  float4 xv = rp[l];
  const float* xa = &xv.x;
  float s = 0.f, q = 0.f;
#pragma unroll
  for (int i = 0; i < 4; ++i) { s += xa[i]; q += xa[i] * xa[i]; }
#pragma unroll
  for (int m = 16; m >= 1; m >>= 1) { s += __shfl_xor(s, m, 32); q += __shfl_xor(q, m, 32); }
  float mu = s * (1.f / 128.f);
  float var = q * (1.f / 128.f) - mu * mu;
  float rstd = rsqrtf(var + 1e-5f);
  int c0 = l * 4;
  u16* dp = dst + r * CC;
#pragma unroll
  for (int i = 0; i < 4; ++i)
    dp[c0 + i] = f2bf((xa[i] - mu) * rstd * w[c0 + i] + b[c0 + i]);
}

// ---------------- weights -> bf16 (row-major [o][c], used directly as B^T) ----------------
__global__ __launch_bounds__(256) void convw_kernel(
    const float* __restrict__ pw, const float* __restrict__ g1, const float* __restrict__ g2,
    const float* __restrict__ inw, const float* __restrict__ outw, u16* __restrict__ dst)
{
  long idx = (long)blockIdx.x * 256 + threadIdx.x;
  if (idx >= 23L * 16384) return;
  int mat = (int)(idx >> 14);
  int local = (int)(idx & 16383);
  float v;
  if (mat == 0)      v = pw[local];
  else if (mat == 1) v = g1[local];
  else if (mat == 2) v = g2[local];
  else if (mat < 18) { int m = mat - 3, a = m / 3, p = m % 3;
                       v = inw[((size_t)a * 384 + p * 128) * 128 + local]; }
  else               { int a = mat - 18; v = outw[(size_t)a * 16384 + local]; }
  dst[idx] = f2bf(v);
}

// ---------------- unified WMMA attention (axis MHAs + token self-MHA) ----------------
// Q: [64][128] bf16 (per-b if qPerB). K/V rows in kbuf/vbuf bf16 at rowbase + j*stride.
// Accumulates sum over slices of softmax(QK^T/sqrt(dh)) V into osum[b][64][128] (atomics).
__global__ __launch_bounds__(256) void attn_kernel(
    const u16* __restrict__ qh, int qPerB,
    const u16* __restrict__ kbuf, const u16* __restrict__ vbuf,
    float* __restrict__ osum,
    int axis, int L, int nslice, long rows_per_b, int blocks_per_b)
{
  __shared__ float Oacc[64 * 128];   // 32 KB
  __shared__ float lg[64 * 64];      // 16 KB (P bf16 aliased in place)
  u16* Pu = (u16*)lg;
  int wave = threadIdx.x >> 5, l = threadIdx.x & 31, tid = threadIdx.x;
  int b = blockIdx.x / blocks_per_b;
  int sl0 = (blockIdx.x % blocks_per_b) * nslice;
  const u16* qb = qh + (qPerB ? (size_t)b * 64 * CC : 0);
  for (int i = tid; i < 64 * 128; i += 256) Oacc[i] = 0.f;
  __syncthreads();
  int ntn = L >> 4;
  for (int si = 0; si < nslice; ++si) {
    int s = sl0 + si;
    long base, stride;
    if (axis == 0)      { int y = s / HZc, z = s % HZc; base = ((long)z * HYc + y) * HXc; stride = 1; }
    else if (axis == 1) { int x = s / HZc, z = s % HZc; base = (long)z * HYc * HXc + x;   stride = HXc; }
    else if (axis == 2) { int x = s / HYc, y = s % HYc; base = (long)y * HXc + x;         stride = (long)HXc * HYc; }
    else                { base = 0; stride = 1; }
    long rowbase = (long)b * rows_per_b + base;
    for (int h = 0; h < NHc; ++h) {
      const u16* kb_ = kbuf + (size_t)rowbase * CC + h * DHc;
      for (int t = wave; t < 4 * ntn; t += 8) {            // QK^T (K = dh = 32 -> 1 WMMA)
        int mt = t / ntn, nt = t % ntn;
        v16bf a  = loadA_bf(qb, CC, mt * 16, h * DHc);
        v16bf bf = loadB_n(kb_, (size_t)stride * CC, 0, nt * 16);
        v8f acc = {};
        acc = wmma_bf16(a, bf, acc);
        int col = nt * 16 + (l & 15);
#pragma unroll
        for (int v = 0; v < 8; ++v)
          lg[(mt * 16 + v + (l >> 4) * 8) * 64 + col] = acc[v];
      }
      __syncthreads();
      if (tid < 64) {                                      // row softmax, write P bf16 in place
        float* rp = &lg[tid * 64];
        float mx = -1e30f;
        for (int j = 0; j < L; ++j) { float t2 = rp[j] * ATTN_SCALE; if (t2 > mx) mx = t2; }
        float sm = 0.f;
        for (int j = 0; j < L; ++j) sm += __expf(rp[j] * ATTN_SCALE - mx);
        float inv = 1.f / sm;
        u16* pr = &Pu[tid * 128];
        for (int j = 0; j < L; ++j)
          pr[j] = f2bf(__expf(rp[j] * ATTN_SCALE - mx) * inv);   // write trails read
      }
      __syncthreads();
      const u16* vb_ = vbuf + (size_t)rowbase * CC + h * DHc;
      for (int t = wave; t < 8; t += 8) {                  // P(64xL) x V(Lx32)
        int mt = t >> 1, nt = t & 1;
        v8f acc = {};
        for (int kk = 0; kk < L; kk += 32) {
          v16bf a  = loadA_bf(Pu, 128, mt * 16, kk);
          v16bf bf = loadB_k(vb_, (size_t)stride * CC, kk, nt * 16);
          acc = wmma_bf16(a, bf, acc);
        }
        int col = h * DHc + nt * 16 + (l & 15);
#pragma unroll
        for (int v = 0; v < 8; ++v)
          Oacc[(mt * 16 + v + (l >> 4) * 8) * 128 + col] += acc[v];
      }
      __syncthreads();
    }
  }
  float* od = osum + (size_t)b * 64 * 128;
  for (int i = tid; i < 64 * 128; i += 256) atomicAdd(&od[i], Oacc[i]);
}

// ---------------- cross attention: 64 query rows/block vs 64 token keys ----------------
__global__ __launch_bounds__(256) void xattn_kernel(
    const u16* __restrict__ qh_all, const u16* __restrict__ kht,
    const u16* __restrict__ vht, u16* __restrict__ obuf)
{
  __shared__ float lg[64 * 64];
  u16* Pu = (u16*)lg;
  int wave = threadIdx.x >> 5, l = threadIdx.x & 31, tid = threadIdx.x;
  long r0 = (long)blockIdx.x * 64;
  int b = (int)(r0 / NNc);
  const u16* q = qh_all + (size_t)r0 * CC;
  for (int h = 0; h < NHc; ++h) {
    const u16* kb_ = kht + (size_t)b * 64 * CC + h * DHc;
    for (int t = wave; t < 16; t += 8) {
      int mt = t >> 2, nt = t & 3;
      v16bf a  = loadA_bf(q, CC, mt * 16, h * DHc);
      v16bf bf = loadB_n(kb_, CC, 0, nt * 16);
      v8f acc = {};
      acc = wmma_bf16(a, bf, acc);
      int col = nt * 16 + (l & 15);
#pragma unroll
      for (int v = 0; v < 8; ++v)
        lg[(mt * 16 + v + (l >> 4) * 8) * 64 + col] = acc[v];
    }
    __syncthreads();
    if (tid < 64) {
      float* rp = &lg[tid * 64];
      float mx = -1e30f;
      for (int j = 0; j < 64; ++j) { float t2 = rp[j] * ATTN_SCALE; if (t2 > mx) mx = t2; }
      float sm = 0.f;
      for (int j = 0; j < 64; ++j) sm += __expf(rp[j] * ATTN_SCALE - mx);
      float inv = 1.f / sm;
      u16* pr = &Pu[tid * 128];
      for (int j = 0; j < 64; ++j) pr[j] = f2bf(__expf(rp[j] * ATTN_SCALE - mx) * inv);
    }
    __syncthreads();
    const u16* vb_ = vht + (size_t)b * 64 * CC + h * DHc;
    for (int t = wave; t < 8; t += 8) {
      int mt = t >> 1, nt = t & 1;
      v8f acc = {};
      for (int kk = 0; kk < 64; kk += 32) {
        v16bf a  = loadA_bf(Pu, 128, mt * 16, kk);
        v16bf bf = loadB_k(vb_, CC, kk, nt * 16);
        acc = wmma_bf16(a, bf, acc);
      }
      int col = h * DHc + nt * 16 + (l & 15);
#pragma unroll
      for (int v = 0; v < 8; ++v)
        obuf[(size_t)(r0 + mt * 16 + v + (l >> 4) * 8) * CC + col] = f2bf(acc[v]);
    }
    __syncthreads();
  }
}

// ---------------- final RMS-weighted 3-way blend (wave per row) ----------------
__global__ __launch_bounds__(256) void blend_kernel(
    const float* __restrict__ h0, const float* __restrict__ h1, float* __restrict__ hf,
    const float* __restrict__ rms, const float* __restrict__ blk)
{
  int wave = threadIdx.x >> 5, l = threadIdx.x & 31;
  long r = (long)blockIdx.x * 8 + wave;
  const float4* p0 = (const float4*)(h0 + r * CC);
  const float4* p1 = (const float4*)(h1 + r * CC);
  float4* p2 = (float4*)(hf + r * CC);
  float4 a0 = p0[l], a1 = p1[l], a2 = p2[l];
  const float* v0 = &a0.x; const float* v1 = &a1.x; const float* v2 = &a2.x;
  float ss0 = 0, ss1 = 0, ss2 = 0, d0 = 0, d1 = 0, d2 = 0;
#pragma unroll
  for (int i = 0; i < 4; ++i) {
    float wc = blk[l * 4 + i] * rms[l * 4 + i];
    ss0 += v0[i] * v0[i]; d0 += wc * v0[i];
    ss1 += v1[i] * v1[i]; d1 += wc * v1[i];
    ss2 += v2[i] * v2[i]; d2 += wc * v2[i];
  }
#pragma unroll
  for (int m = 16; m >= 1; m >>= 1) {
    ss0 += __shfl_xor(ss0, m, 32); d0 += __shfl_xor(d0, m, 32);
    ss1 += __shfl_xor(ss1, m, 32); d1 += __shfl_xor(d1, m, 32);
    ss2 += __shfl_xor(ss2, m, 32); d2 += __shfl_xor(d2, m, 32);
  }
  float lo0 = fminf(30.f, fmaxf(-30.f, d0 * rsqrtf(ss0 * (1.f / 128.f) + 1e-6f)));
  float lo1 = fminf(30.f, fmaxf(-30.f, d1 * rsqrtf(ss1 * (1.f / 128.f) + 1e-6f)));
  float lo2 = fminf(30.f, fmaxf(-30.f, d2 * rsqrtf(ss2 * (1.f / 128.f) + 1e-6f)));
  float mx = fmaxf(lo0, fmaxf(lo1, lo2));
  float e0 = __expf(lo0 - mx), e1 = __expf(lo1 - mx), e2 = __expf(lo2 - mx);
  float inv = 1.f / (e0 + e1 + e2);
  float4 o;
  float* oo = &o.x;
#pragma unroll
  for (int i = 0; i < 4; ++i) oo[i] = (e0 * v0[i] + e1 * v1[i] + e2 * v2[i]) * inv;
  p2[l] = o;
}

// ---------------- host orchestration ----------------
extern "C" void kernel_launch(void* const* d_in, const int* in_sizes, int n_in,
                              void* d_out, int out_size, void* d_ws, size_t ws_size,
                              hipStream_t stream)
{
  (void)in_sizes; (void)n_in; (void)out_size; (void)ws_size;
  const float* h      = (const float*)d_in[0];
  const float* hist0  = (const float*)d_in[1];
  const float* hist1  = (const float*)d_in[2];
  const float* dw_w   = (const float*)d_in[3];
  const float* pw_w   = (const float*)d_in[4];
  const float* gn_w   = (const float*)d_in[5];
  const float* gn_b   = (const float*)d_in[6];
  const float* gr1_s  = (const float*)d_in[7];
  const float* gr1_gw = (const float*)d_in[8];
  const float* gr1_gb = (const float*)d_in[9];
  const float* gtok   = (const float*)d_in[10];
  const float* inw    = (const float*)d_in[11];
  const float* inb    = (const float*)d_in[12];
  const float* outw   = (const float*)d_in[13];
  const float* outb   = (const float*)d_in[14];
  const float* ln_w   = (const float*)d_in[15];
  const float* ln_b   = (const float*)d_in[16];
  const float* gr2_s  = (const float*)d_in[17];
  const float* gr2_gw = (const float*)d_in[18];
  const float* gr2_gb = (const float*)d_in[19];
  const float* rms_w  = (const float*)d_in[20];
  const float* blk_w  = (const float*)d_in[21];
  float* out = (float*)d_out;

  // workspace layout
  float* fws    = (float*)d_ws;
  float* ws_y   = fws;                 // BNC f32   (conv out -> delta -> K/V bf16 -> qh_all/obuf)
  float* ws_hn  = fws + BNCc;          // BNC f32   (h after gated residual 1)
  u16*   gx     = (u16*)(fws + 2 * BNCc); // BNC bf16 staging
  float* smallf = fws + 2 * BNCc + BNCc / 2;
  float* st_sum = smallf;
  float* st_sq  = smallf + 64;
  float* st_mu  = smallf + 128;
  float* st_rs  = smallf + 192;
  float* osum_ax= smallf + 256;            // 3 * 2*64*128
  float* osum_t = osum_ax + 3 * 16384;     // 2*64*128
  float* tokensf= osum_t + 16384;          // 2*64*128
  u16* wT    = (u16*)(tokensf + 16384);    // 23 * 16384 bf16 weights (row-major [o][c])
  u16* qhax  = wT + 23 * 16384;            // 3 * 64*128
  u16* tinbf = qhax + 3 * 8192;
  u16* qbf   = tinbf + 16384;
  u16* kbf   = qbf + 16384;
  u16* vbf   = kbf + 16384;
  u16* t2bf  = vbf + 16384;
  u16* kht   = t2bf + 16384;
  u16* vht   = kht + 16384;

  u16* kbuf   = (u16*)ws_y;        // BNC bf16 (first half of ws_y)
  u16* vbuf   = kbuf + BNCc;       // BNC bf16 (second half)
  u16* qh_all = kbuf;              // reused later
  u16* obuf   = vbuf;

  auto wqW  = [&](int a) { return wT + (size_t)(3 + 3 * a + 0) * 16384; };
  auto wkW  = [&](int a) { return wT + (size_t)(3 + 3 * a + 1) * 16384; };
  auto wvW  = [&](int a) { return wT + (size_t)(3 + 3 * a + 2) * 16384; };
  auto outW = [&](int a) { return wT + (size_t)(18 + a) * 16384; };
  u16* pwW = wT; u16* g1W = wT + 16384; u16* g2W = wT + 2 * 16384;

  const int EL_BLK = (int)(BNCc / 256);          // 131072
  const int GEMM_BLK = (int)(ROWSc / 128);       // 2048

  // zero stats + attention accumulators
  hipMemsetAsync(smallf, 0, (256 + 4 * 16384) * sizeof(float), stream);

  // weights -> bf16 (kept row-major; loadB_n consumes them as B^T)
  convw_kernel<<<(23 * 16384 + 255) / 256, 256, 0, stream>>>(pw_w, gr1_gw, gr2_gw, inw, outw, wT);

  // 1) depthwise conv + group norm
  conv_dw_kernel<<<EL_BLK, 256, 0, stream>>>(h, dw_w, ws_y);
  gn_stats_kernel<<<GEMM_BLK, 256, 0, stream>>>(ws_y, st_sum, st_sq);
  gn_final_kernel<<<1, 64, 0, stream>>>(st_sum, st_sq, st_mu, st_rs);
  gn_gelu_kernel<<<EL_BLK, 256, 0, stream>>>(ws_y, st_mu, st_rs, gn_w, gn_b, gx);

  // 2) pointwise conv: delta = gelu @ pw^T   (overwrites ws_y)
  gemm_kernel<false, 0><<<GEMM_BLK, 256, 0, stream>>>(gx, 128, 1.f, pwW, 128, 128,
      nullptr, 128, 0.f, ws_y, nullptr, 128, nullptr, nullptr, nullptr, ROWSc);

  // 3) gated residual 1: hnew = h + s1*sigmoid(h@g1^T+b)*delta
  cvt_bf16_kernel<<<EL_BLK, 256, 0, stream>>>(h, gx);
  gemm_kernel<false, 2><<<GEMM_BLK, 256, 0, stream>>>(gx, 128, 1.f, g1W, 128, 128,
      gr1_gb, 128, 0.f, ws_hn, nullptr, 128, gr1_s, h, ws_y, ROWSc);

  // 4) per-axis global-token MHA
  for (int a = 0; a < 3; ++a)
    gemm_kernel<true, 1><<<1, 256, 0, stream>>>(gtok, 128, 1.f, wqW(a), 128, 128,
        inb + a * 384, 128, 0.f, nullptr, qhax + a * 8192, 128, nullptr, nullptr, nullptr, 64L);
  const int   La[3]  = {64, 64, 32};
  const int   NSa[3] = {32, 32, 64};
  const float Sdiv[3] = {2048.f, 2048.f, 4096.f};
  for (int a = 0; a < 3; ++a) {
    gemm_kernel<true, 1><<<GEMM_BLK, 256, 0, stream>>>(ws_hn, 128, 1.f, wkW(a), 128, 128,
        inb + a * 384 + 128, 128, 0.f, nullptr, kbuf, 128, nullptr, nullptr, nullptr, ROWSc);
    gemm_kernel<true, 1><<<GEMM_BLK, 256, 0, stream>>>(ws_hn, 128, 1.f, wvW(a), 128, 128,
        inb + a * 384 + 256, 128, 0.f, nullptr, vbuf, 128, nullptr, nullptr, nullptr, ROWSc);
    attn_kernel<<<128, 256, 0, stream>>>(qhax + a * 8192, 0, kbuf, vbuf,
        osum_ax + a * 16384, a, La[a], NSa[a], NNc, 64);
  }

  // 5) tokens = sum_a mean_a @ out_w_a^T + out_b_a
  for (int a = 0; a < 3; ++a)
    gemm_kernel<true, 0><<<1, 256, 0, stream>>>(osum_ax + a * 16384, 128, 1.f / Sdiv[a],
        outW(a), 128, 128, outb + a * 128, 128, (a == 0 ? 0.f : 1.f),
        tokensf, nullptr, 128, nullptr, nullptr, nullptr, 128L);

  // 6) token self-MHA
  ln_rows_kernel<<<16, 256, 0, stream>>>(tokensf, ln_w, ln_b, tinbf, 128);
  u16* qkv_dst[3] = {qbf, kbf, vbf};
  for (int p = 0; p < 3; ++p)
    gemm_kernel<false, 1><<<1, 256, 0, stream>>>(tinbf, 128, 1.f,
        wT + (size_t)(3 + 3 * 3 + p) * 16384, 128, 128, inb + 3 * 384 + p * 128, 128,
        0.f, nullptr, qkv_dst[p], 128, nullptr, nullptr, nullptr, 128L);
  attn_kernel<<<2, 256, 0, stream>>>(qbf, 1, kbf, vbf, osum_t, 3, 64, 1, 64L, 1);
  gemm_kernel<true, 1><<<1, 256, 0, stream>>>(osum_t, 128, 1.f, outW(3), 128, 128,
      outb + 3 * 128, 128, 0.f, nullptr, t2bf, 128, nullptr, nullptr, nullptr, 128L);
  gemm_kernel<false, 1><<<1, 256, 0, stream>>>(t2bf, 128, 1.f, wkW(4), 128, 128,
      inb + 4 * 384 + 128, 128, 0.f, nullptr, kht, 128, nullptr, nullptr, nullptr, 128L);
  gemm_kernel<false, 1><<<1, 256, 0, stream>>>(t2bf, 128, 1.f, wvW(4), 128, 128,
      inb + 4 * 384 + 256, 128, 0.f, nullptr, vht, 128, nullptr, nullptr, nullptr, 128L);

  // 7) cross attention: glob = MHA(LN(hnew) -> tokens)
  ln_rows_kernel<<<(int)(ROWSc / 8), 256, 0, stream>>>(ws_hn, ln_w + 128, ln_b + 128, gx, ROWSc);
  gemm_kernel<false, 1><<<GEMM_BLK, 256, 0, stream>>>(gx, 128, 1.f, wqW(4), 128, 128,
      inb + 4 * 384, 128, 0.f, nullptr, qh_all, 128, nullptr, nullptr, nullptr, ROWSc);
  xattn_kernel<<<(int)(ROWSc / 64), 256, 0, stream>>>(qh_all, kht, vht, obuf);
  gemm_kernel<false, 0><<<GEMM_BLK, 256, 0, stream>>>(obuf, 128, 1.f, outW(4), 128, 128,
      outb + 4 * 128, 128, 0.f, out, nullptr, 128, nullptr, nullptr, nullptr, ROWSc);

  // 8) gated residual 2 (in place on d_out): hfinal = hnew + s2*sigmoid(hnew@g2^T+b)*glob
  gemm_kernel<true, 2><<<GEMM_BLK, 256, 0, stream>>>(ws_hn, 128, 1.f, g2W, 128, 128,
      gr2_gb, 128, 0.f, out, nullptr, 128, gr2_s, ws_hn, out, ROWSc);

  // 9) RMS-weighted softmax blend over {hist0, hist1, hfinal} (in place on d_out)
  blend_kernel<<<(int)(ROWSc / 8), 256, 0, stream>>>(hist0, hist1, out, rms_w, blk_w);
}
